// RelationNetworks_21002390078047
// MI455X (gfx1250) — matrix operand order repaired and checked by reading, not compiled, not run on hardware
//
#include <hip/hip_runtime.h>
#include <hip/hip_bf16.h>
#include <math.h>

typedef _Float16 h16;
typedef __attribute__((ext_vector_type(16))) _Float16 v16h_t;
typedef __attribute__((ext_vector_type(8)))  _Float16 v8h_t;
typedef __attribute__((ext_vector_type(8)))  float    v8f_t;

#define G_AS   264   // LDS activation row stride in halfs (>=256, 4-bank step => conflict-free)
#define G_ROWS 32    // pair-rows per workgroup

// ---------------- conv (3x3, stride 2, pad 1) ----------------
__global__ void k_conv3x3s2(const float* __restrict__ x, const float* __restrict__ w,
                            float* __restrict__ y,
                            int N, int Ci, int Co, int Hi, int Wi, int Ho, int Wo)
{
    int idx = blockIdx.x * blockDim.x + threadIdx.x;
    int total = N * Co * Ho * Wo;
    if (idx >= total) return;
    int wo = idx % Wo; int t = idx / Wo;
    int ho = t % Ho;   t /= Ho;
    int co = t % Co;   int n = t / Co;
    float acc = 0.0f;
    for (int ci = 0; ci < Ci; ++ci) {
        const float* xp = x + ((size_t)(n * Ci + ci) * Hi) * Wi;
        const float* wp = w + ((size_t)(co * Ci + ci) * 3) * 3;
        #pragma unroll
        for (int kh = 0; kh < 3; ++kh) {
            int hi = ho * 2 - 1 + kh;
            if (hi < 0 || hi >= Hi) continue;
            #pragma unroll
            for (int kw = 0; kw < 3; ++kw) {
                int wi = wo * 2 - 1 + kw;
                if (wi < 0 || wi >= Wi) continue;
                acc += xp[hi * Wi + wi] * wp[kh * 3 + kw];
            }
        }
    }
    y[idx] = acc;
}

// ---------------- BN training stats: mean/var per channel ----------------
__global__ void k_bnstats(const float* __restrict__ y, float* __restrict__ stats,
                          int N, int C, int H, int W)
{
    int c = blockIdx.x;
    int tid = threadIdx.x;
    int per = H * W;
    int cnt = N * per;
    float s1 = 0.0f, s2 = 0.0f;
    for (int i = tid; i < cnt; i += blockDim.x) {
        int n = i / per; int r = i - n * per;
        float v = y[((size_t)(n * C + c)) * per + r];
        s1 += v; s2 += v * v;
    }
    __shared__ float r1[256], r2[256];
    r1[tid] = s1; r2[tid] = s2;
    __syncthreads();
    for (int s = 128; s > 0; s >>= 1) {
        if (tid < s) { r1[tid] += r1[tid + s]; r2[tid] += r2[tid + s]; }
        __syncthreads();
    }
    if (tid == 0) {
        float m = r1[0] / (float)cnt;
        float v = r2[0] / (float)cnt - m * m;
        stats[c * 2 + 0] = m;
        stats[c * 2 + 1] = v;
    }
}

// ---------------- BN apply + ReLU (in place) ----------------
__global__ void k_bnrelu(float* __restrict__ y, const float* __restrict__ stats,
                         const float* __restrict__ g, const float* __restrict__ b,
                         int total, int C, int HW)
{
    int idx = blockIdx.x * blockDim.x + threadIdx.x;
    if (idx >= total) return;
    int c = (idx / HW) % C;
    float m = stats[c * 2 + 0], v = stats[c * 2 + 1];
    float val = (y[idx] - m) * rsqrtf(v + 1e-5f) * g[c] + b[c];
    y[idx] = val > 0.0f ? val : 0.0f;
}

// ---------------- LSTM (single block: B=64, T=32, E=32, H=128) ----------------
__global__ __launch_bounds__(1024) void k_lstm(
    const float* __restrict__ emb, const int* __restrict__ question,
    const int* __restrict__ qlen,
    const float* __restrict__ w_ih, const float* __restrict__ w_hh,
    const float* __restrict__ b_ih, const float* __restrict__ b_hh,
    float* __restrict__ hq)
{
    __shared__ float hs[8192];   // [64][128]
    __shared__ float xt[2048];   // [64][32]
    int tid = threadIdx.x;
    float cr[8];                 // cell state, private per owned cell
    for (int i = tid; i < 8192; i += 1024) hs[i] = 0.0f;
    #pragma unroll
    for (int u = 0; u < 8; ++u) cr[u] = 0.0f;
    __syncthreads();

    for (int t = 0; t < 32; ++t) {
        for (int i = tid; i < 2048; i += 1024) {
            int b = i >> 5, e = i & 31;
            int tok = question[b * 32 + t];
            xt[i] = emb[tok * 32 + e];
        }
        __syncthreads();
        float gv[8][4];
        for (int u = 0; u < 8; ++u) {
            int cell = tid + u * 1024;
            int b = cell >> 7, j = cell & 127;
            #pragma unroll
            for (int gi = 0; gi < 4; ++gi) {
                int row = gi * 128 + j;
                float acc = b_ih[row] + b_hh[row];
                const float* wi = w_ih + (size_t)row * 32;
                const float* xp = xt + b * 32;
                for (int e = 0; e < 32; ++e) acc += xp[e] * wi[e];
                const float* wh = w_hh + (size_t)row * 128;
                const float* hp = hs + b * 128;
                for (int k = 0; k < 128; ++k) acc += hp[k] * wh[k];
                gv[u][gi] = acc;
            }
        }
        __syncthreads();
        for (int u = 0; u < 8; ++u) {
            int cell = tid + u * 1024;
            int b = cell >> 7;
            float ig = 1.0f / (1.0f + expf(-gv[u][0]));
            float fg = 1.0f / (1.0f + expf(-gv[u][1]));
            float gg = tanhf(gv[u][2]);
            float og = 1.0f / (1.0f + expf(-gv[u][3]));
            float cn = fg * cr[u] + ig * gg;
            float hn = og * tanhf(cn);
            cr[u] = cn;
            hs[cell] = hn;
            if (t == qlen[b] - 1) hq[cell] = hn;
        }
        __syncthreads();
    }
}

// ---------------- weight prep: f32 [K][N] -> f16 WT[N][Kpad] (zero-padded K) ----------------
__global__ void k_wprep(const float* __restrict__ src, h16* __restrict__ dst,
                        int K, int N, int Kpad)
{
    int idx = blockIdx.x * blockDim.x + threadIdx.x;
    if (idx >= N * Kpad) return;
    int n = idx / Kpad, k = idx - n * Kpad;
    dst[idx] = (k < K) ? (h16)src[(size_t)k * N + n] : (h16)0.0f;
}

__global__ void k_zero(float* __restrict__ p, int n)
{
    int i = blockIdx.x * blockDim.x + threadIdx.x;
    if (i < n) p[i] = 0.0f;
}

// ---------------- fused 4-layer g-MLP with on-chip pair-sum ----------------
__device__ __forceinline__ v16h_t ld_frag16(const h16* p0, const h16* p1)
{
    v8h_t lo = *(const v8h_t*)p0;
    v8h_t hi = *(const v8h_t*)p1;
    v16h_t r;
    #pragma unroll
    for (int i = 0; i < 8; ++i) { r[i] = lo[i]; r[i + 8] = hi[i]; }
    return r;
}

// Wave computes a 32x32 output tile: 2 row-blocks x 2 n-tiles.
// Each A/B fragment feeds 2 WMMAs; next k-step fragments are loaded before
// issuing the current k-step's WMMAs (software double-buffer).
__device__ __forceinline__ void g_layer(const h16* actIn, const h16* __restrict__ wT,
                                        const int Kt, const int Kpad,
                                        int lane, int nbase, v8f_t c[2][2])
{
    const int l16   = lane >> 4;
    const int koffA = l16 << 3;   // lanes 0-15: K {0..7,16..23}; 16-31: K {8..15,24..31}
    const int koffB = l16 << 3;   // lanes 0-15: K {0..7,16..23}; 16-31: K {8..15,24..31}
    const int lc    = lane & 15;
    const h16* aB0 = actIn + lc * G_AS + koffA;          // rows 0..15
    const h16* aB1 = actIn + (16 + lc) * G_AS + koffA;   // rows 16..31
    const h16* bB0 = wT + (size_t)(nbase + lc) * Kpad + koffB;
    const h16* bB1 = wT + (size_t)(nbase + 16 + lc) * Kpad + koffB;

    v16h_t a0 = ld_frag16(aB0, aB0 + 16);
    v16h_t a1 = ld_frag16(aB1, aB1 + 16);
    v16h_t b0 = ld_frag16(bB0, bB0 + 16);
    v16h_t b1 = ld_frag16(bB1, bB1 + 16);
    #pragma unroll
    for (int kt = 0; kt < Kt; ++kt) {
        v16h_t na0 = a0, na1 = a1, nb0 = b0, nb1 = b1;
        if (kt + 1 < Kt) {
            const int kb = (kt + 1) * 32;
            na0 = ld_frag16(aB0 + kb, aB0 + kb + 16);
            na1 = ld_frag16(aB1 + kb, aB1 + kb + 16);
            nb0 = ld_frag16(bB0 + kb, bB0 + kb + 16);
            nb1 = ld_frag16(bB1 + kb, bB1 + kb + 16);
        }
        c[0][0] = __builtin_amdgcn_wmma_f32_16x16x32_f16(false, a0, false, b0, (short)0, c[0][0], false, false);
        c[0][1] = __builtin_amdgcn_wmma_f32_16x16x32_f16(false, a0, false, b1, (short)0, c[0][1], false, false);
        c[1][0] = __builtin_amdgcn_wmma_f32_16x16x32_f16(false, a1, false, b0, (short)0, c[1][0], false, false);
        c[1][1] = __builtin_amdgcn_wmma_f32_16x16x32_f16(false, a1, false, b1, (short)0, c[1][1], false, false);
        a0 = na0; a1 = na1; b0 = nb0; b1 = nb1;
    }
}

__device__ __forceinline__ void g_store2(h16* actOut, const float* __restrict__ bias,
                                         int lane, int nbase, v8f_t c[2][2])
{
    const int lc    = lane & 15;
    const int rhalf = (lane >> 4) << 3;   // lanes 0-15: rows m=e; 16-31: m=8+e
    #pragma unroll
    for (int nti = 0; nti < 2; ++nti) {
        const int col = nbase + nti * 16 + lc;
        const float bb = bias[col];
        #pragma unroll
        for (int rbi = 0; rbi < 2; ++rbi) {
            const int rbase = rbi * 16 + rhalf;
            #pragma unroll
            for (int e = 0; e < 8; ++e) {
                float d = c[rbi][nti][e] + bb;
                d = d > 0.0f ? d : 0.0f;
                actOut[(rbase + e) * G_AS + col] = (h16)d;
            }
        }
    }
}

__global__ __launch_bounds__(256) void k_g_fused(
    const float* __restrict__ x4,   // [64,24,8,8] post conv4 bn-relu
    const float* __restrict__ hq,   // [64,128]
    const h16* __restrict__ wT1, const h16* __restrict__ wT2,
    const h16* __restrict__ wT3, const h16* __restrict__ wT4,
    const float* __restrict__ gb1, const float* __restrict__ gb2,
    const float* __restrict__ gb3,
    float* __restrict__ gsum)       // [64,256] f32, atomically accumulated
{
    __shared__ h16 A0[G_ROWS * G_AS];
    __shared__ h16 A1[G_ROWS * G_AS];
    const int b       = blockIdx.x >> 7;    // 128 row-tiles per batch (4096/32)
    const int tile    = blockIdx.x & 127;
    const int rowBase = tile * G_ROWS;
    const int tid     = threadIdx.x;

    // Build layer-1 input tile (f16): [cv[q](24) | cv[p](24) | hq(128) | zero-pad to 192]
    for (int idx = tid; idx < G_ROWS * 192; idx += 256) {
        const int r = idx / 192, k = idx - r * 192;
        const int pidx = rowBase + r;
        const int p = pidx >> 6, q = pidx & 63;
        float v;
        if (k < 24)       v = x4[(b * 24 + k) * 64 + q];
        else if (k < 48)  v = x4[(b * 24 + (k - 24)) * 64 + p];
        else if (k < 176) v = hq[b * 128 + (k - 48)];
        else              v = 0.0f;
        A0[r * G_AS + k] = (h16)v;
    }
    __syncthreads();

    const int wid  = tid >> 5, lane = tid & 31;
    const int nbase = wid * 32;    // 8 waves x 32 columns = 256

    v8f_t c[2][2];
    const v8f_t zero = {0.f, 0.f, 0.f, 0.f, 0.f, 0.f, 0.f, 0.f};

    // layer 1: K=176 (padded 192)
    #pragma unroll
    for (int i = 0; i < 2; ++i) { c[i][0] = zero; c[i][1] = zero; }
    g_layer(A0, wT1, 6, 192, lane, nbase, c);
    g_store2(A1, gb1, lane, nbase, c);
    __syncthreads();
    // layer 2
    #pragma unroll
    for (int i = 0; i < 2; ++i) { c[i][0] = zero; c[i][1] = zero; }
    g_layer(A1, wT2, 8, 256, lane, nbase, c);
    g_store2(A0, gb2, lane, nbase, c);
    __syncthreads();
    // layer 3
    #pragma unroll
    for (int i = 0; i < 2; ++i) { c[i][0] = zero; c[i][1] = zero; }
    g_layer(A0, wT3, 8, 256, lane, nbase, c);
    g_store2(A1, gb3, lane, nbase, c);
    __syncthreads();
    // layer 4: no bias/relu here (4096*g_b4 folded into f-MLP); reduce rows on-chip
    #pragma unroll
    for (int i = 0; i < 2; ++i) { c[i][0] = zero; c[i][1] = zero; }
    g_layer(A1, wT4, 8, 256, lane, nbase, c);
    #pragma unroll
    for (int nti = 0; nti < 2; ++nti) {
        float s = 0.0f;
        #pragma unroll
        for (int rbi = 0; rbi < 2; ++rbi)
            #pragma unroll
            for (int e = 0; e < 8; ++e) s += c[rbi][nti][e];
        s += __shfl_xor(s, 16, 32);            // combine rows m<8 with m>=8 (same column)
        if ((lane & 16) == 0)
            atomicAdd(&gsum[b * 256 + nbase + nti * 16 + (lane & 15)], s);
    }
}

// ---------------- small FC (f-MLP), fp32 ----------------
__global__ void k_fc(const float* __restrict__ in, const float* __restrict__ w,
                     const float* __restrict__ bias,
                     const float* __restrict__ foldBias, float foldScale,
                     float* __restrict__ out, int M, int K, int N, int relu)
{
    int idx = blockIdx.x * blockDim.x + threadIdx.x;
    if (idx >= M * N) return;
    int m = idx / N, n = idx - m * N;
    float acc = bias[n];
    for (int k = 0; k < K; ++k) {
        float x = in[(size_t)m * K + k];
        if (foldBias) x += foldScale * foldBias[k];
        acc += x * w[(size_t)k * N + n];
    }
    if (relu && acc < 0.0f) acc = 0.0f;
    out[idx] = acc;
}

// ---------------- launch ----------------
extern "C" void kernel_launch(void* const* d_in, const int* in_sizes, int n_in,
                              void* d_out, int out_size, void* d_ws, size_t ws_size,
                              hipStream_t stream)
{
    (void)in_sizes; (void)n_in; (void)out_size; (void)ws_size;
    const float* image    = (const float*)d_in[0];
    const int*   question = (const int*)d_in[1];
    const int*   qlen     = (const int*)d_in[2];
    const float* cw[4]  = {(const float*)d_in[3], (const float*)d_in[6],
                           (const float*)d_in[9], (const float*)d_in[12]};
    const float* bng[4] = {(const float*)d_in[4], (const float*)d_in[7],
                           (const float*)d_in[10], (const float*)d_in[13]};
    const float* bnb[4] = {(const float*)d_in[5], (const float*)d_in[8],
                           (const float*)d_in[11], (const float*)d_in[14]};
    const float* emb  = (const float*)d_in[15];
    const float* w_ih = (const float*)d_in[16];
    const float* w_hh = (const float*)d_in[17];
    const float* b_ih = (const float*)d_in[18];
    const float* b_hh = (const float*)d_in[19];
    const float* g_w[4] = {(const float*)d_in[20], (const float*)d_in[22],
                           (const float*)d_in[24], (const float*)d_in[26]};
    const float* g_b[4] = {(const float*)d_in[21], (const float*)d_in[23],
                           (const float*)d_in[25], (const float*)d_in[27]};
    const float* f_w1 = (const float*)d_in[28];
    const float* f_b1 = (const float*)d_in[29];
    const float* f_w2 = (const float*)d_in[30];
    const float* f_b2 = (const float*)d_in[31];
    const float* f_w3 = (const float*)d_in[32];
    const float* f_b3 = (const float*)d_in[33];
    float* out = (float*)d_out;

    char* ws = (char*)d_ws;
    size_t off = 0;
    auto take = [&](size_t bytes) -> char* {
        char* p = ws + off;
        off += (bytes + 255) & ~(size_t)255;
        return p;
    };
    float* y1 = (float*)take((size_t)64 * 24 * 64 * 64 * 4);
    float* y2 = (float*)take((size_t)64 * 24 * 32 * 32 * 4);
    float* y3 = (float*)take((size_t)64 * 24 * 16 * 16 * 4);
    float* y4 = (float*)take((size_t)64 * 24 * 8 * 8 * 4);
    float* st[4];
    for (int i = 0; i < 4; ++i) st[i] = (float*)take(24 * 2 * 4);
    float* hq   = (float*)take(64 * 128 * 4);
    h16*   wT1  = (h16*)take(256 * 192 * 2);
    h16*   wT2  = (h16*)take(256 * 256 * 2);
    h16*   wT3  = (h16*)take(256 * 256 * 2);
    h16*   wT4  = (h16*)take(256 * 256 * 2);
    float* gsum = (float*)take(64 * 256 * 4);
    float* fb1  = (float*)take(64 * 256 * 4);
    float* fb2  = (float*)take(64 * 256 * 4);

    // conv stack
    {
        int tot = 64 * 24 * 64 * 64;
        k_conv3x3s2<<<(tot + 255) / 256, 256, 0, stream>>>(image, cw[0], y1, 64, 3, 24, 128, 128, 64, 64);
        k_bnstats<<<24, 256, 0, stream>>>(y1, st[0], 64, 24, 64, 64);
        k_bnrelu<<<(tot + 255) / 256, 256, 0, stream>>>(y1, st[0], bng[0], bnb[0], tot, 24, 64 * 64);
    }
    {
        int tot = 64 * 24 * 32 * 32;
        k_conv3x3s2<<<(tot + 255) / 256, 256, 0, stream>>>(y1, cw[1], y2, 64, 24, 24, 64, 64, 32, 32);
        k_bnstats<<<24, 256, 0, stream>>>(y2, st[1], 64, 24, 32, 32);
        k_bnrelu<<<(tot + 255) / 256, 256, 0, stream>>>(y2, st[1], bng[1], bnb[1], tot, 24, 32 * 32);
    }
    {
        int tot = 64 * 24 * 16 * 16;
        k_conv3x3s2<<<(tot + 255) / 256, 256, 0, stream>>>(y2, cw[2], y3, 64, 24, 24, 32, 32, 16, 16);
        k_bnstats<<<24, 256, 0, stream>>>(y3, st[2], 64, 24, 16, 16);
        k_bnrelu<<<(tot + 255) / 256, 256, 0, stream>>>(y3, st[2], bng[2], bnb[2], tot, 24, 16 * 16);
    }
    {
        int tot = 64 * 24 * 8 * 8;
        k_conv3x3s2<<<(tot + 255) / 256, 256, 0, stream>>>(y3, cw[3], y4, 64, 24, 24, 16, 16, 8, 8);
        k_bnstats<<<24, 256, 0, stream>>>(y4, st[3], 64, 24, 8, 8);
        k_bnrelu<<<(tot + 255) / 256, 256, 0, stream>>>(y4, st[3], bng[3], bnb[3], tot, 24, 8 * 8);
    }

    // LSTM
    k_lstm<<<1, 1024, 0, stream>>>(emb, question, qlen, w_ih, w_hh, b_ih, b_hh, hq);

    // weight prep + gsum init
    k_wprep<<<(256 * 192 + 255) / 256, 256, 0, stream>>>(g_w[0], wT1, 176, 256, 192);
    k_wprep<<<(256 * 256 + 255) / 256, 256, 0, stream>>>(g_w[1], wT2, 256, 256, 256);
    k_wprep<<<(256 * 256 + 255) / 256, 256, 0, stream>>>(g_w[2], wT3, 256, 256, 256);
    k_wprep<<<(256 * 256 + 255) / 256, 256, 0, stream>>>(g_w[3], wT4, 256, 256, 256);
    k_zero<<<(64 * 256 + 255) / 256, 256, 0, stream>>>(gsum, 64 * 256);

    // fused relation MLP: 64 batches x 128 tiles of 32 pair-rows
    k_g_fused<<<64 * 128, 256, 0, stream>>>(y4, hq, wT1, wT2, wT3, wT4,
                                            g_b[0], g_b[1], g_b[2], gsum);

    // f-MLP (fold 4096*g_b4 into the first layer's input)
    k_fc<<<(64 * 256 + 255) / 256, 256, 0, stream>>>(gsum, f_w1, f_b1, g_b[3], 4096.0f,
                                                     fb1, 64, 256, 256, 1);
    k_fc<<<(64 * 256 + 255) / 256, 256, 0, stream>>>(fb1, f_w2, f_b2, (const float*)nullptr, 0.0f,
                                                     fb2, 64, 256, 256, 1);
    k_fc<<<(64 * 29 + 255) / 256, 256, 0, stream>>>(fb2, f_w3, f_b3, (const float*)nullptr, 0.0f,
                                                    out, 64, 256, 29, 0);
}
